// STMM_RFCN_8675833938742
// MI455X (gfx1250) — matrix operand order repaired and checked by reading, not compiled, not run on hardware
//
#include <hip/hip_runtime.h>
#include <hip/hip_bf16.h>

typedef __attribute__((ext_vector_type(16))) _Float16 v16h;
typedef __attribute__((ext_vector_type(8)))  _Float16 v8h;
typedef __attribute__((ext_vector_type(4)))  _Float16 v4h;
typedef __attribute__((ext_vector_type(8)))  float    v8f;

#define KKP   7
#define NCLSC 31
#define NBAT  4
#define NRPB  128
#define NROI  512                 // NBAT*NRPB
#define CH    1024
#define HH    40
#define WWD   40
#define NBIN  49                  // KKP*KKP
#define KDIM  (NBIN*CH)           // 50176
#define OPAD  160                 // 155 outputs padded to 160
#define NSEG  8
#define SEGK  (KDIM/NSEG)         // 6272 = 196*32
#define SCALEF (1.0f/32.0f)

// ---------------- 1) feat (B,C,H,W) -> featT (B,H,W,C) fp32 ----------------
__global__ __launch_bounds__(256) void k_transpose(const float* __restrict__ feat,
                                                   float* __restrict__ featT) {
    int idx = blockIdx.x * 256 + threadIdx.x;     // NBAT*HH*WWD*CH = 6,553,600 exact
    int c   = idx & (CH - 1);
    int pix = idx >> 10;                          // b*1600 + h*40 + w
    int b   = pix / (HH * WWD);
    int hw  = pix - b * (HH * WWD);
    featT[idx] = feat[(b * CH + c) * (HH * WWD) + hw];
}

// ---------------- 2) pack weights -> Wh (OPAD, KDIM) f16, rows 155..159 = 0 ---
__global__ __launch_bounds__(256) void k_packw(const float* __restrict__ wcls,
                                               const float* __restrict__ wbbox,
                                               _Float16* __restrict__ Wh) {
    int idx = blockIdx.x * 256 + threadIdx.x;     // OPAD*KDIM = 8,028,160 exact
    int o = idx / KDIM;
    int k = idx - o * KDIM;
    float v = 0.0f;
    if (o < NCLSC)      v = wcls[o * KDIM + k];               // (31,49,1024) row-major
    else if (o < 155)   v = wbbox[(o - NCLSC) * KDIM + k];    // (124,49,1024) row-major
    Wh[idx] = (_Float16)v;
}

// ---------------- 3) PSRoI pooling -> G (NROI, KDIM) f16, scale folded --------
__global__ __launch_bounds__(256) void k_pool(const float* __restrict__ featT,
                                              const float* __restrict__ rois,
                                              _Float16* __restrict__ G) {
    int idx = blockIdx.x;             // n*49 + bin
    int n   = idx / NBIN;
    int bin = idx - n * NBIN;
    int ph  = bin / KKP;
    int pw  = bin - ph * KKP;
    int b   = n >> 7;                 // n / NRPB

    const float* roi = rois + n * 5;
    float x1 = rintf(roi[1]) * SCALEF;
    float y1 = rintf(roi[2]) * SCALEF;
    float x2 = rintf(roi[3] + 1.0f) * SCALEF;
    float y2 = rintf(roi[4] + 1.0f) * SCALEF;
    float rw = fmaxf(x2 - x1, 0.1f);
    float rh = fmaxf(y2 - y1, 0.1f);
    float bh = rh / 7.0f;
    float bw = rw / 7.0f;
    int hs = (int)fminf(fmaxf(floorf((float)ph * bh + y1), 0.0f), (float)HH);
    int he = (int)fminf(fmaxf(ceilf(((float)ph + 1.0f) * bh + y1), 0.0f), (float)HH);
    int wsx = (int)fminf(fmaxf(floorf((float)pw * bw + x1), 0.0f), (float)WWD);
    int wex = (int)fminf(fmaxf(ceilf(((float)pw + 1.0f) * bw + x1), 0.0f), (float)WWD);
    int cnt = (he - hs) * (wex - wsx);
    float sc = (cnt > 0) ? (1.0f / (49.0f * (float)cnt)) : 0.0f;

    int c0 = threadIdx.x * 4;         // 256 threads * 4 ch = 1024
    float a0 = 0.f, a1 = 0.f, a2 = 0.f, a3 = 0.f;
    for (int h = hs; h < he; ++h) {
        const float* row = featT + (((b * HH + h) * WWD) << 10) + c0;
        for (int w = wsx; w < wex; ++w) {
            const float4 v = *(const float4*)(row + (w << 10));
            a0 += v.x; a1 += v.y; a2 += v.z; a3 += v.w;
        }
    }
    v4h r;
    r[0] = (_Float16)(a0 * sc); r[1] = (_Float16)(a1 * sc);
    r[2] = (_Float16)(a2 * sc); r[3] = (_Float16)(a3 * sc);
    *(v4h*)(G + (size_t)idx * 1024 + c0) = r;     // row n, col bin*1024+c
}

// ---------------- 4) WMMA GEMM: part[s] += G(512,K) x Wh(160,K)^T -------------
__global__ __launch_bounds__(256) void k_gemm(const _Float16* __restrict__ G,
                                              const _Float16* __restrict__ Wh,
                                              float* __restrict__ part) {
    int wv    = blockIdx.x * 8 + (threadIdx.x >> 5);   // 0..2559
    int seg   = wv / 320;                              // K segment
    int tile  = wv - seg * 320;                        // 32 x 10 tiles
    int tileM = tile / 10;
    int tileN = tile - tileM * 10;
    int lane  = threadIdx.x & 31;
    int half  = lane >> 4;
    int l16   = lane & 15;

    const _Float16* aBase = G  + (size_t)(tileM * 16 + l16) * KDIM;
    const _Float16* bBase = Wh + (size_t)(tileN * 16 + l16) * KDIM;
    int k0 = seg * SEGK;

    v8f acc = {};
    #pragma unroll 2
    for (int k = k0; k < k0 + SEGK; k += 32) {
        // A 16x32 f16: lanes0-15 hold K=k..k+15, lanes16-31 hold K=k+16..k+31 (contiguous)
        v16h A = *(const v16h*)(aBase + k + half * 16);
        // B 32x16 f16: lanes0-15 hold K=k..k+15, lanes16-31 hold K=k+16..k+31 (contiguous)
        v16h Bv = *(const v16h*)(bBase + k + half * 16);
        acc = __builtin_amdgcn_wmma_f32_16x16x32_f16(
            false, A, false, Bv, (short)0, acc, false, false);
    }
    // D layout: VGPR i -> M = i + half*8, N = l16
    float* dst = part + (size_t)seg * NROI * OPAD;
    int n = tileN * 16 + l16;
    #pragma unroll
    for (int i = 0; i < 8; ++i) {
        int m = tileM * 16 + half * 8 + i;
        dst[m * OPAD + n] = acc[i];
    }
}

// ---------------- 5) per-ROI softmax / gather / smooth-L1 ---------------------
__global__ __launch_bounds__(32) void k_post(const float* __restrict__ part,
                                             const int* __restrict__ label,
                                             const float* __restrict__ target,
                                             const float* __restrict__ inw,
                                             const float* __restrict__ outw,
                                             float* __restrict__ out,
                                             float* __restrict__ lossA,
                                             float* __restrict__ lossB) {
    int n = blockIdx.x;          // 0..511
    int o = threadIdx.x;         // 0..31
    float logit = -3.0e38f;
    if (o < NCLSC) {
        float s = 0.f;
        #pragma unroll
        for (int g = 0; g < NSEG; ++g) s += part[g * NROI * OPAD + n * OPAD + o];
        logit = s;
    }
    float m = logit;
    for (int off = 16; off; off >>= 1) m = fmaxf(m, __shfl_xor(m, off, 32));
    float e = (o < NCLSC) ? expf(logit - m) : 0.0f;
    float sum = e;
    for (int off = 16; off; off >>= 1) sum += __shfl_xor(sum, off, 32);
    if (o < NCLSC) out[n * NCLSC + o] = e / sum;

    int lab = label[n];
    float lsum = logf(sum);
    float logit_lab = __shfl(logit, lab, 32);
    if (o == 0) lossA[n] = -(logit_lab - m - lsum);

    if (o < 4) {
        float sel = 0.f;
        int col = NCLSC + lab * 4 + o;
        #pragma unroll
        for (int g = 0; g < NSEG; ++g) sel += part[g * NROI * OPAD + n * OPAD + col];
        out[NROI * NCLSC + n * 4 + o] = sel;
        float diff = inw[n * 4 + o] * (sel - target[n * 4 + o]);
        float ad = fabsf(diff);
        float il = (ad < 1.0f) ? 0.5f * diff * diff : (ad - 0.5f);
        float v = outw[n * 4 + o] * il;
        v += __shfl_xor(v, 1, 32);
        v += __shfl_xor(v, 2, 32);
        if (o == 0) lossB[n] = v;
    }
}

// ---------------- 6) deterministic loss reduction -----------------------------
__global__ __launch_bounds__(256) void k_loss(const float* __restrict__ lossA,
                                              const float* __restrict__ lossB,
                                              float* __restrict__ out) {
    __shared__ float sa[256], sb[256];
    int t = threadIdx.x;
    sa[t] = lossA[t] + lossA[t + 256];
    sb[t] = lossB[t] + lossB[t + 256];
    __syncthreads();
    for (int s = 128; s; s >>= 1) {
        if (t < s) { sa[t] += sa[t + s]; sb[t] += sb[t + s]; }
        __syncthreads();
    }
    if (t == 0) {
        out[NROI * NCLSC + NROI * 4 + 0] = sa[0] / (float)NROI;
        out[NROI * NCLSC + NROI * 4 + 1] = sb[0] / (float)NROI;
    }
}

extern "C" void kernel_launch(void* const* d_in, const int* in_sizes, int n_in,
                              void* d_out, int out_size, void* d_ws, size_t ws_size,
                              hipStream_t stream) {
    const float* feat   = (const float*)d_in[0];
    const float* rois   = (const float*)d_in[1];
    const int*   label  = (const int*)d_in[2];
    const float* target = (const float*)d_in[3];
    const float* inw    = (const float*)d_in[4];
    const float* outw   = (const float*)d_in[5];
    const float* wcls   = (const float*)d_in[6];
    const float* wbbox  = (const float*)d_in[7];
    float* out = (float*)d_out;

    char* ws = (char*)d_ws;
    float*    featT = (float*)(ws);                        // 26,214,400 B
    _Float16* G     = (_Float16*)(ws + 26214400);          // 51,380,224 B
    _Float16* Wh    = (_Float16*)(ws + 77594624);          // 16,056,320 B
    float*    part  = (float*)(ws + 93650944);             //  2,621,440 B
    float*    lossA = (float*)(ws + 96272384);             //      2,048 B
    float*    lossB = (float*)(ws + 96274432);             //      2,048 B

    k_transpose<<<25600, 256, 0, stream>>>(feat, featT);
    k_packw   <<<31360, 256, 0, stream>>>(wcls, wbbox, Wh);
    k_pool    <<<NROI * NBIN, 256, 0, stream>>>(featT, rois, G);
    k_gemm    <<<320, 256, 0, stream>>>(G, Wh, part);
    k_post    <<<NROI, 32, 0, stream>>>(part, label, target, inw, outw, out, lossA, lossB);
    k_loss    <<<1, 256, 0, stream>>>(lossA, lossB, out);
}
